// BaseLSTM_68496138437439
// MI455X (gfx1250) — matrix operand order repaired and checked by reading, not compiled, not run on hardware
//
#include <hip/hip_runtime.h>

typedef __attribute__((ext_vector_type(16))) _Float16 v16h;
typedef __attribute__((ext_vector_type(8)))  _Float16 v8h;
typedef __attribute__((ext_vector_type(8)))  float    v8f;
typedef __attribute__((ext_vector_type(4)))  float    v4f;

#define BB 64
#define SS 2048
#define DD 64
#define HH 128
#define NK 6          // K = 192 = 6 * 32
#define NW 8          // waves per workgroup

__device__ __forceinline__ float fsig(float x) {
  return 1.0f / (1.0f + __expf(-x));
}
__device__ __forceinline__ float ftanh(float x) {
  x = fminf(fmaxf(x, -20.0f), 20.0f);
  const float e = __expf(2.0f * x);
  return (e - 1.0f) / (e + 1.0f);
}

// One workgroup = 16 batch rows, persistent over all 2048 timesteps.
// Wave n computes gate tiles {n, n+8, n+16, n+24} of z = [x_t|h] @ [Wx;Wh] + b,
// i.e. i/f/g/o for h-columns 16n..16n+15 -> gate math is wave-local.
__global__ __launch_bounds__(256, 1)
void lstm_wmma_persistent(const float* __restrict__ x,  const float* __restrict__ Wx,
                          const float* __restrict__ Wh, const float* __restrict__ b,
                          const float* __restrict__ Wd, const float* __restrict__ bd,
                          float* __restrict__ out)
{
  __shared__ __align__(16) _Float16 hbuf[2][16][HH];  // double-buffered h (f16), 8 KB
  __shared__ float pbuf[2][NW][16];                   // head partials, 1 KB

  const int tid  = threadIdx.x;
  const int wav  = tid >> 5;          // 0..7
  const int lane = tid & 31;
  const int lh   = lane >> 4;         // half-wave: 0 or 1
  const int lm   = lane & 15;
  const int row0 = blockIdx.x * 16;   // batch rows [row0, row0+16)
  const int col  = wav * 16 + lm;     // h-column owned by this lane in C tiles

  // ---- one-time: build register-resident B fragments (f16) from Wx;Wh ----
  // B-frag layout (32x16 f16): lane element e -> K = kk*32 + lh*8 + (e<8 ? e : e+8), N = col
  v16h Bf[4][NK];
  #pragma unroll
  for (int g = 0; g < 4; ++g) {
    const int n = g * HH + col;       // column of z for gate g
    #pragma unroll
    for (int kk = 0; kk < NK; ++kk) {
      #pragma unroll
      for (int e = 0; e < 16; ++e) {
        const int k = kk * 32 + lh * 8 + (e < 8 ? e : e + 8);
        const float w = (k < DD) ? Wx[(size_t)k * (4 * HH) + n]
                                 : Wh[(size_t)(k - DD) * (4 * HH) + n];
        Bf[g][kk][e] = (_Float16)w;
      }
    }
  }

  // per-lane constants: gate biases for this column, head weight
  const float bgate[4] = { b[col], b[col + HH], b[col + 2 * HH], b[col + 3 * HH] };
  const float wd  = Wd[col];
  const float bd0 = bd[0];

  // cell state for this lane's 8 (row, col) elements
  float c[8];
  #pragma unroll
  for (int v = 0; v < 8; ++v) c[v] = 0.0f;

  // h(0) = 0
  for (int i = tid; i < 16 * HH; i += 256) (&hbuf[0][0][0])[i] = (_Float16)0.0f;
  __syncthreads();

  // A-fragment addressing: lane feeds row M = lm; K chunks at lh*8 and lh*8+16
  const int arow = lm;
  const float* xrow = x + (size_t)(row0 + arow) * SS * DD + lh * 8;

  // ---- software pipeline: pre-convert x A-fragments for t = 0 ----
  v16h xa[2];
  #pragma unroll
  for (int kk = 0; kk < 2; ++kk) {
    const float* p = xrow + kk * 32;
    const v4f x0 = *(const v4f*)(p);
    const v4f x1 = *(const v4f*)(p + 4);
    const v4f x2 = *(const v4f*)(p + 16);
    const v4f x3 = *(const v4f*)(p + 20);
    #pragma unroll
    for (int e = 0; e < 4; ++e) {
      xa[kk][e]      = (_Float16)x0[e];
      xa[kk][e + 4]  = (_Float16)x1[e];
      xa[kk][e + 8]  = (_Float16)x2[e];
      xa[kk][e + 12] = (_Float16)x3[e];
    }
  }

  for (int t = 0; t < SS; ++t) {
    const int rb = t & 1;
    const int wb = rb ^ 1;

    // ---- issue ALL h A-fragment LDS loads up front (distinct registers) ----
    v8h hlo[4], hhi[4];
    #pragma unroll
    for (int kk = 0; kk < 4; ++kk) {
      const int hc = kk * 32 + lh * 8;          // h-column base for slice kk
      hlo[kk] = *(const v8h*)(&hbuf[rb][arow][hc]);
      hhi[kk] = *(const v8h*)(&hbuf[rb][arow][hc + 16]);
    }

    // ---- issue global prefetch of x for t+1 (uniform, branch-free) ----
    const int tn = (t + 1 < SS) ? (t + 1) : t;
    const float* pn = xrow + (size_t)tn * DD;
    const v4f nx[8] = {
      *(const v4f*)(pn),      *(const v4f*)(pn + 4),
      *(const v4f*)(pn + 16), *(const v4f*)(pn + 20),
      *(const v4f*)(pn + 32), *(const v4f*)(pn + 36),
      *(const v4f*)(pn + 48), *(const v4f*)(pn + 52)
    };

    // ---- accumulators seeded with the per-column gate bias ----
    v8f acc[4];
    #pragma unroll
    for (int g = 0; g < 4; ++g) {
      #pragma unroll
      for (int v = 0; v < 8; ++v) acc[g][v] = bgate[g];
    }

    // K-slices 0..1 from x_t (already in registers -> covers LDS latency)
    #pragma unroll
    for (int kk = 0; kk < 2; ++kk) {
      #pragma unroll
      for (int g = 0; g < 4; ++g) {
        acc[g] = __builtin_amdgcn_wmma_f32_16x16x32_f16(
            false, xa[kk], false, Bf[g][kk], (short)0, acc[g], false, false);
      }
    }
    // K-slices 2..5 from h(t)
    #pragma unroll
    for (int kk = 0; kk < 4; ++kk) {
      v16h a;
      #pragma unroll
      for (int e = 0; e < 8; ++e) { a[e] = hlo[kk][e]; a[e + 8] = hhi[kk][e]; }
      #pragma unroll
      for (int g = 0; g < 4; ++g) {
        acc[g] = __builtin_amdgcn_wmma_f32_16x16x32_f16(
            false, a, false, Bf[g][kk + 2], (short)0, acc[g], false, false);
      }
    }

    // ---- gates + state update (acc element v ~ row v + 8*lh, col 'col') ----
    float part[8];
    #pragma unroll
    for (int v = 0; v < 8; ++v) {
      const float iz = fsig(acc[0][v]);
      const float fz = fsig(acc[1][v]);
      const float gz = ftanh(acc[2][v]);
      const float oz = fsig(acc[3][v]);
      const float cn = fz * c[v] + iz * gz;
      c[v] = cn;
      const float hn = oz * ftanh(cn);
      part[v] = hn * wd;                         // head partial
      hbuf[wb][v + 8 * lh][col] = (_Float16)hn;  // publish h(t+1)
    }

    // ---- convert prefetched x for next step (loads have had WMMA+gate cover) --
    #pragma unroll
    for (int kk = 0; kk < 2; ++kk) {
      #pragma unroll
      for (int e = 0; e < 4; ++e) {
        xa[kk][e]      = (_Float16)nx[4 * kk + 0][e];
        xa[kk][e + 4]  = (_Float16)nx[4 * kk + 1][e];
        xa[kk][e + 8]  = (_Float16)nx[4 * kk + 2][e];
        xa[kk][e + 12] = (_Float16)nx[4 * kk + 3][e];
      }
    }

    // ---- head partial: reduce over this wave's 16 columns (per 16-lane half) --
    #pragma unroll
    for (int v = 0; v < 8; ++v) {
      part[v] += __shfl_xor(part[v], 1, 32);
      part[v] += __shfl_xor(part[v], 2, 32);
      part[v] += __shfl_xor(part[v], 4, 32);
      part[v] += __shfl_xor(part[v], 8, 32);
    }
    if (lm == 0) {
      #pragma unroll
      for (int v = 0; v < 8; ++v) pbuf[wb][wav][8 * lh + v] = part[v];
    }

    __syncthreads();  // h(t+1) + partials visible; single barrier per step

    // wave 0 finishes the Dense(1, sigmoid) head for step t
    if (wav == 0 && lane < 16) {
      float s = bd0;
      #pragma unroll
      for (int w = 0; w < NW; ++w) s += pbuf[wb][w][lane];
      out[(size_t)(row0 + lane) * SS + t] = fsig(s);
    }
  }
}

extern "C" void kernel_launch(void* const* d_in, const int* in_sizes, int n_in,
                              void* d_out, int out_size, void* d_ws, size_t ws_size,
                              hipStream_t stream) {
  (void)in_sizes; (void)n_in; (void)d_ws; (void)ws_size; (void)out_size;
  const float* x  = (const float*)d_in[0];
  const float* Wx = (const float*)d_in[1];
  const float* Wh = (const float*)d_in[2];
  const float* b  = (const float*)d_in[3];
  const float* Wd = (const float*)d_in[4];
  const float* bd = (const float*)d_in[5];
  float* out = (float*)d_out;

  dim3 grid(BB / 16);   // 4 independent workgroups (batch-split recurrences)
  dim3 block(256);      // 8 waves of 32
  hipLaunchKernelGGL(lstm_wmma_persistent, grid, block, 0, stream,
                     x, Wx, Wh, b, Wd, bd, out);
}
